// ViDAInjectedLNormLayer_42975442763795
// MI455X (gfx1250) — matrix operand order repaired and checked by reading, not compiled, not run on hardware
//
#include <hip/hip_runtime.h>
#include <hip/hip_bf16.h>
#include <math.h>

#define Bc 8
#define Nc 2048
#define Cc 768
#define Hc 128
#define POOLc 64
#define SEGPOOL 160   // positions per pool along L = 5*N = 10240

typedef __attribute__((ext_vector_type(16))) _Float16 v16h;
typedef __attribute__((ext_vector_type(8)))  float    v8f;

// ---------- helpers ----------
__device__ inline float blockReduceSum256(float v, float* sm) {
    int tid = threadIdx.x;
    sm[tid] = v; __syncthreads();
    for (int s = 128; s > 0; s >>= 1) {
        if (tid < s) sm[tid] += sm[tid + s];
        __syncthreads();
    }
    float r = sm[0]; __syncthreads();
    return r;
}

__device__ inline float seg_overlap(int lo, int hi, int a, int bnd) {
    int v = min(hi, bnd) - max(lo, a);
    return v > 0 ? (float)v : 0.0f;
}

__device__ inline uint32_t rotl32(uint32_t x, int r) { return (x << r) | (x >> (32 - r)); }

// JAX threefry2x32 with key = jax.random.key(1234) -> (k0,k1)=(0,1234)
__device__ inline void threefry2x32(uint32_t x0, uint32_t x1, uint32_t& o0, uint32_t& o1) {
    const uint32_t k0 = 0u, k1 = 1234u;
    uint32_t ks[3] = { k0, k1, k0 ^ k1 ^ 0x1BD11BDAu };
    x0 += ks[0]; x1 += ks[1];
    const int R[2][4] = { {13, 15, 26, 6}, {17, 29, 16, 24} };
    for (int g = 0; g < 5; ++g) {
        const int* rr = R[g & 1];
        for (int i = 0; i < 4; ++i) { x0 += x1; x1 = rotl32(x1, rr[i]); x1 ^= x0; }
        x0 += ks[(g + 1) % 3];
        x1 += ks[(g + 2) % 3] + (uint32_t)(g + 1);
    }
    o0 = x0; o1 = x1;
}

__device__ inline float gumbel_from_index(int j) {
    // bits for flat index j of shape (768,2): counts split as x0=0..767, x1=768..1535
    uint32_t o0, o1, bits;
    if (j < 768) { threefry2x32((uint32_t)j, (uint32_t)(j + 768), o0, o1); bits = o0; }
    else         { threefry2x32((uint32_t)(j - 768), (uint32_t)j, o0, o1); bits = o1; }
    uint32_t fb = (bits >> 9) | 0x3F800000u;
    float u01 = __uint_as_float(fb) - 1.0f;          // [0,1)
    float u = fmaxf(u01 * (1.0f - 1e-10f) + 1e-10f, 1e-10f);
    return -logf(-logf(u));
}

// ---------- K1: per-batch column means of x and x2 ----------
__global__ void k1_colmean(const float* __restrict__ x, const float* __restrict__ x2,
                           float* __restrict__ am1, float* __restrict__ am2) {
    int b = blockIdx.x / 3, chunk = blockIdx.x % 3;
    int c = chunk * 256 + threadIdx.x;
    float s1 = 0.f, s2 = 0.f;
    for (int n = 0; n < Nc; ++n) {
        size_t idx = ((size_t)(b * Nc + n)) * Cc + c;
        s1 += x[idx]; s2 += x2[idx];
    }
    am1[b * Cc + c] = s1 * (1.0f / Nc);
    am2[b * Cc + c] = s2 * (1.0f / Nc);
}

// ---------- K1b: norms of the mean vectors ----------
__global__ void k1b_norms(const float* __restrict__ am1, const float* __restrict__ am2,
                          float* __restrict__ namv) {
    __shared__ float sm[256];
    int b = blockIdx.x, tid = threadIdx.x;
    float q1 = 0.f, q2 = 0.f;
    for (int c = tid; c < Cc; c += 256) {
        float v1 = am1[b * Cc + c], v2 = am2[b * Cc + c];
        q1 += v1 * v1; q2 += v2 * v2;
    }
    q1 = blockReduceSum256(q1, sm);
    q2 = blockReduceSum256(q2, sm);
    if (tid == 0) { namv[b] = sqrtf(q1); namv[8 + b] = sqrtf(q2); }
}

// ---------- K2: per-token raw cosine similarities (one wave32 per token) ----------
__global__ void k2_csim(const float* __restrict__ x, const float* __restrict__ x2,
                        const float* __restrict__ am1, const float* __restrict__ am2,
                        const float* __restrict__ namv,
                        float* __restrict__ csim1, float* __restrict__ csim2) {
    int t = blockIdx.x * 8 + (threadIdx.x >> 5);  // token id in [0, B*N)
    int lane = threadIdx.x & 31;
    int b = t >> 11;                               // N = 2048
    float d1 = 0.f, ss2 = 0.f, d2 = 0.f, ss1 = 0.f;
    size_t base = (size_t)t * Cc;
    int amoff = b * Cc;
    for (int c = lane; c < Cc; c += 32) {
        float xv = x[base + c], yv = x2[base + c];
        float a1v = am1[amoff + c], a2v = am2[amoff + c];
        d1 += yv * a1v; ss2 += yv * yv;   // csim1: dot(x2, mean(x)), ||x2||
        d2 += xv * a2v; ss1 += xv * xv;   // csim2: dot(x,  mean(x2)), ||x||
    }
    for (int m = 16; m > 0; m >>= 1) {
        d1 += __shfl_xor(d1, m, 32); ss2 += __shfl_xor(ss2, m, 32);
        d2 += __shfl_xor(d2, m, 32); ss1 += __shfl_xor(ss1, m, 32);
    }
    if (lane == 0) {
        csim1[t] = d1 / (namv[b] * sqrtf(ss2));
        csim2[t] = d2 / (namv[8 + b] * sqrtf(ss1));
    }
}

// ---------- K3: per-batch min/max/sum -> coef scalars ----------
// coef[b,n] = (csim - c0) / (sum - N*c0),  c0 = cmin if cmax>cmin else 0
__global__ void k3_stats(const float* __restrict__ csim1, const float* __restrict__ csim2,
                         float* __restrict__ stats) {
    __shared__ float smin[256], smax[256], ssum[256];
    int b = blockIdx.x, tid = threadIdx.x;
    for (int which = 0; which < 2; ++which) {
        const float* cs = which ? csim2 : csim1;
        float mn = 3.4e38f, mx = -3.4e38f, s = 0.f;
        for (int n = tid; n < Nc; n += 256) {
            float v = cs[b * Nc + n];
            mn = fminf(mn, v); mx = fmaxf(mx, v); s += v;
        }
        smin[tid] = mn; smax[tid] = mx; ssum[tid] = s; __syncthreads();
        for (int st = 128; st > 0; st >>= 1) {
            if (tid < st) {
                smin[tid] = fminf(smin[tid], smin[tid + st]);
                smax[tid] = fmaxf(smax[tid], smax[tid + st]);
                ssum[tid] += ssum[tid + st];
            }
            __syncthreads();
        }
        if (tid == 0) {
            float cmin = smin[0], cmax = smax[0], S = ssum[0];
            float c0 = (cmax > cmin) ? cmin : 0.0f;
            stats[which * 16 + b]     = c0;
            stats[which * 16 + 8 + b] = 1.0f / (S - (float)Nc * c0);
        }
        __syncthreads();
    }
}

// ---------- K4: per-pool weighted row sums  V3/V2/V1 (64 x 768) ----------
// segment k = globalpos>>11: 0:x->p3W, 1:x2->p3W, 2:c1->p2W, 3:c2->p2W, 4:(x-x2)->p1W
__global__ void k4_poolsums(const float* __restrict__ x, const float* __restrict__ x2,
                            const float* __restrict__ csim1, const float* __restrict__ csim2,
                            const float* __restrict__ stats,
                            float* __restrict__ V3, float* __restrict__ V2, float* __restrict__ V1) {
    __shared__ float coef[SEGPOOL];
    int p = blockIdx.x, j = threadIdx.x;   // 768 threads = channels
    float a3 = 0.f, a2 = 0.f, a1 = 0.f;
    for (int b = 0; b < Bc; ++b) {
        if (j < SEGPOOL) {
            int gp = p * SEGPOOL + j;
            int k = gp >> 11, n = gp & (Nc - 1);
            float cv = 1.0f;
            if (k == 2)      cv = (csim1[b * Nc + n] - stats[b])      * stats[8 + b];
            else if (k == 3) cv = (csim2[b * Nc + n] - stats[16 + b]) * stats[24 + b];
            coef[j] = cv;
        }
        __syncthreads();
        for (int t = 0; t < SEGPOOL; ++t) {
            int gp = p * SEGPOOL + t;
            int k = gp >> 11, n = gp & (Nc - 1);
            size_t base = ((size_t)(b * Nc + n)) * Cc + j;
            if (k == 0)      a3 += x[base];
            else if (k == 1) a3 += x2[base];
            else if (k == 2) a2 += x[base] * coef[t];
            else if (k == 3) a2 += x2[base] * coef[t];
            else             a1 += x[base] - x2[base];
        }
        __syncthreads();
    }
    const float sc = 1.0f / (SEGPOOL * Bc);   // /160 pool mean, /8 batch mean
    V3[p * Cc + j] = a3 * sc;
    V2[p * Cc + j] = a2 * sc;
    V1[p * Cc + j] = a1 * sc;
}

// ---------- K5 (WMMA): logit[c,p] = (V3@p3W + V2@p2W + V1@p1W)[p,c] + bias ----------
// One wave per 16x16 tile. grid = (4 pool-blocks, 48 channel-blocks).
__global__ void k5_logit_wmma(const float* __restrict__ V3, const float* __restrict__ V2,
                              const float* __restrict__ V1,
                              const float* __restrict__ p3W, const float* __restrict__ p2W,
                              const float* __restrict__ p1W,
                              const float* __restrict__ p3b, const float* __restrict__ p2b,
                              const float* __restrict__ p1b,
                              float* __restrict__ logit) {
    int lane  = threadIdx.x;            // 0..31
    int m0    = blockIdx.x * 16;        // pool block
    int n0    = blockIdx.y * 16;        // channel block
    int row   = lane & 15;
    int khalf = (lane >> 4) << 4;       // A: +16 K for upper half-lanes
    int kb    = (lane >> 4) << 4;       // B: +16 K for upper half-lanes
    v8f acc = {};
    #pragma unroll
    for (int t = 0; t < 3; ++t) {
        const float* V = (t == 0) ? V3 : (t == 1) ? V2 : V1;
        const float* W = (t == 0) ? p3W : (t == 1) ? p2W : p1W;
        for (int k = 0; k < Cc; k += 32) {
            v16h a, bm;
            #pragma unroll
            for (int i = 0; i < 16; ++i) {
                // A 16x32 f16 layout: element i holds K=khalf+i (contiguous, matches B)
                int ka = khalf + i;
                a[i]  = (_Float16)V[(m0 + row) * Cc + k + ka];
                // B 32x16 f16 layout: lane half selects K range, lane&15 selects column
                bm[i] = (_Float16)W[(size_t)(k + kb + i) * Cc + n0 + row];
            }
            acc = __builtin_amdgcn_wmma_f32_16x16x32_f16(false, a, false, bm,
                                                         (short)0, acc, false, false);
        }
    }
    // D layout: VGPR r -> pool m0 + r + 8*(lane>>4), channel n0 + (lane&15)
    int c = n0 + row;
    #pragma unroll
    for (int r = 0; r < 8; ++r) {
        int p = m0 + r + ((lane >> 4) << 3);
        int lo = p * SEGPOOL, hi = lo + SEGPOOL;
        float cnt3 = seg_overlap(lo, hi, 0,      2 * Nc);
        float cnt2 = seg_overlap(lo, hi, 2 * Nc, 4 * Nc);
        float cnt1 = seg_overlap(lo, hi, 4 * Nc, 5 * Nc);
        float bias = (cnt3 * p3b[c] + cnt2 * p2b[c] + cnt1 * p1b[c]) * (1.0f / SEGPOOL);
        logit[c * POOLc + p] = acc[r] + bias;
    }
}

// ---------- K6: both routers: sigmoid(relu(z@W1+b1)@W2+b2), z = logit (768,64) ----------
__global__ void k6_router(const float* __restrict__ logit,
                          const float* __restrict__ W1a, const float* __restrict__ b1a,
                          const float* __restrict__ W2a, const float* __restrict__ b2a,
                          const float* __restrict__ W1b, const float* __restrict__ b1b,
                          const float* __restrict__ W2b, const float* __restrict__ b2b,
                          float* __restrict__ scale, float* __restrict__ sr) {
    __shared__ float zs[32 * 64];
    __shared__ float hs[32 * 128];
    int c0 = blockIdx.x * 32, tid = threadIdx.x;
    for (int u = tid; u < 32 * 64; u += 256) {
        int r = u >> 6, p = u & 63;
        zs[u] = logit[(c0 + r) * 64 + p];
    }
    __syncthreads();
    for (int which = 0; which < 2; ++which) {
        const float* W1 = which ? W1b : W1a; const float* b1 = which ? b1b : b1a;
        const float* W2 = which ? W2b : W2a; const float* b2 = which ? b2b : b2a;
        float* out = which ? sr : scale;
        for (int u = tid; u < 32 * 128; u += 256) {
            int r = u >> 7, hh = u & 127;
            float s = b1[hh];
            for (int p = 0; p < 64; ++p) s += zs[r * 64 + p] * W1[p * 128 + hh];
            hs[u] = fmaxf(s, 0.f);
        }
        __syncthreads();
        if (tid < 64) {
            int r = tid >> 1, o = tid & 1;
            float s = b2[o];
            for (int hh = 0; hh < 128; ++hh) s += hs[r * 128 + hh] * W2[hh * 2 + o];
            out[(c0 + r) * 2 + o] = 1.0f / (1.0f + expf(-s));
        }
        __syncthreads();
    }
}

// ---------- K7: Gumbel straight-through + modulated LN weight vectors ----------
__global__ void k7_modweights(const float* __restrict__ scale, const float* __restrict__ sr,
                              const float* __restrict__ ln_w, const float* __restrict__ ln_b,
                              const float* __restrict__ sw1, const float* __restrict__ sb1,
                              const float* __restrict__ sw2, const float* __restrict__ sb2,
                              float* __restrict__ wvec) {
    int c = blockIdx.x * 256 + threadIdx.x;
    if (c >= Cc) return;
    float a0 = sr[c * 2 + 0] + gumbel_from_index(c * 2 + 0);
    float a1 = sr[c * 2 + 1] + gumbel_from_index(c * 2 + 1);
    float m = fmaxf(a0, a1);
    float e0 = expf(a0 - m), e1 = expf(a1 - m);
    float ys0 = e0 / (e0 + e1);
    float hard0 = (a0 >= a1) ? 1.0f : 0.0f;       // argmax ties -> first index
    float shift = (hard0 + ys0) - ys0;            // straight-through forward value
    float s0 = scale[c * 2 + 0], s1 = scale[c * 2 + 1];
    wvec[c]        = ln_w[c] * (1.f + s0) + shift * sw1[c];
    wvec[768 + c]  = ln_b[c] * (1.f + s1) + shift * sb1[c];
    wvec[1536 + c] = ln_w[c] * (1.f + s0) + shift * sw2[c];
    wvec[2304 + c] = ln_b[c] * (1.f + s1) + shift * sb2[c];
}

// ---------- K8: fused dual layernorm (ddof=1) ----------
__global__ void k8_ln(const float* __restrict__ x, const float* __restrict__ x2,
                      const float* __restrict__ wvec,
                      float* __restrict__ out1, float* __restrict__ out2) {
    __shared__ float sm[256];
    int t = blockIdx.x, tid = threadIdx.x;
    size_t base = (size_t)t * Cc;
    float xv[3], yv[3];
    float s1 = 0.f, q1 = 0.f, s2 = 0.f, q2 = 0.f;
    #pragma unroll
    for (int i = 0; i < 3; ++i) {
        int c = tid + i * 256;
        xv[i] = x[base + c]; yv[i] = x2[base + c];
        s1 += xv[i]; q1 += xv[i] * xv[i];
        s2 += yv[i]; q2 += yv[i] * yv[i];
    }
    s1 = blockReduceSum256(s1, sm);
    q1 = blockReduceSum256(q1, sm);
    s2 = blockReduceSum256(s2, sm);
    q2 = blockReduceSum256(q2, sm);
    float m1 = s1 * (1.0f / Cc);
    float sd1 = sqrtf(fmaxf(q1 - (float)Cc * m1 * m1, 0.f) * (1.0f / (Cc - 1)));
    float m2 = s2 * (1.0f / Cc);
    float sd2 = sqrtf(fmaxf(q2 - (float)Cc * m2 * m2, 0.f) * (1.0f / (Cc - 1)));
    float i1 = 1.f / (sd1 + 1e-10f), i2 = 1.f / (sd2 + 1e-10f);
    #pragma unroll
    for (int i = 0; i < 3; ++i) {
        int c = tid + i * 256;
        float w1 = wvec[c], b1 = wvec[768 + c], w2 = wvec[1536 + c], b2 = wvec[2304 + c];
        out1[base + c] = w1 * (xv[i] - m1) * i1 + b1;
        out2[base + c] = w2 * (yv[i] - m2) * i2 + b2;
    }
}

// ---------- launch ----------
extern "C" void kernel_launch(void* const* d_in, const int* in_sizes, int n_in,
                              void* d_out, int out_size, void* d_ws, size_t ws_size,
                              hipStream_t stream) {
    (void)in_sizes; (void)n_in; (void)out_size; (void)ws_size;
    const float* x    = (const float*)d_in[0];
    const float* x2   = (const float*)d_in[1];
    const float* ln_w = (const float*)d_in[2];
    const float* ln_b = (const float*)d_in[3];
    const float* sw1  = (const float*)d_in[4];
    const float* sb1  = (const float*)d_in[5];
    const float* sw2  = (const float*)d_in[6];
    const float* sb2  = (const float*)d_in[7];
    const float* p1W  = (const float*)d_in[8];
    const float* p1b  = (const float*)d_in[9];
    const float* p2W  = (const float*)d_in[10];
    const float* p2b  = (const float*)d_in[11];
    const float* p3W  = (const float*)d_in[12];
    const float* p3b  = (const float*)d_in[13];
    const float* r1W1 = (const float*)d_in[14];
    const float* r1b1 = (const float*)d_in[15];
    const float* r1W2 = (const float*)d_in[16];
    const float* r1b2 = (const float*)d_in[17];
    const float* r2W1 = (const float*)d_in[18];
    const float* r2b1 = (const float*)d_in[19];
    const float* r2W2 = (const float*)d_in[20];
    const float* r2b2 = (const float*)d_in[21];

    float* out1 = (float*)d_out;
    float* out2 = out1 + (size_t)Bc * Nc * Cc;

    float* ws    = (float*)d_ws;
    float* am1   = ws;                 // 6144
    float* am2   = am1   + Bc * Cc;    // 6144
    float* namv  = am2   + Bc * Cc;    // 16
    float* csim1 = namv  + 16;         // 16384
    float* csim2 = csim1 + Bc * Nc;    // 16384
    float* stats = csim2 + Bc * Nc;    // 32
    float* V3    = stats + 32;         // 49152
    float* V2    = V3    + POOLc * Cc; // 49152
    float* V1    = V2    + POOLc * Cc; // 49152
    float* logit = V1    + POOLc * Cc; // 49152
    float* scale = logit + Cc * POOLc; // 1536
    float* sr    = scale + Cc * 2;     // 1536
    float* wvec  = sr    + Cc * 2;     // 3072

    k1_colmean <<<dim3(Bc * 3), 256, 0, stream>>>(x, x2, am1, am2);
    k1b_norms  <<<dim3(Bc),     256, 0, stream>>>(am1, am2, namv);
    k2_csim    <<<dim3((Bc * Nc) / 8), 256, 0, stream>>>(x, x2, am1, am2, namv, csim1, csim2);
    k3_stats   <<<dim3(Bc),     256, 0, stream>>>(csim1, csim2, stats);
    k4_poolsums<<<dim3(POOLc),  768, 0, stream>>>(x, x2, csim1, csim2, stats, V3, V2, V1);
    k5_logit_wmma<<<dim3(4, 48), 32, 0, stream>>>(V3, V2, V1, p3W, p2W, p1W, p3b, p2b, p1b, logit);
    k6_router  <<<dim3(Cc / 32), 256, 0, stream>>>(logit, r1W1, r1b1, r1W2, r1b2,
                                                   r2W1, r2b1, r2W2, r2b2, scale, sr);
    k7_modweights<<<dim3(3), 256, 0, stream>>>(scale, sr, ln_w, ln_b, sw1, sb1, sw2, sb2, wvec);
    k8_ln      <<<dim3(Bc * Nc), 256, 0, stream>>>(x, x2, wvec, out1, out2);
}
